// MoeFFN_88974542504099
// MI455X (gfx1250) — matrix operand order repaired and checked by reading, not compiled, not run on hardware
//
#include <hip/hip_runtime.h>
#include <hip/hip_bf16.h>
#include <math.h>

typedef __bf16 v16bf __attribute__((ext_vector_type(16)));
typedef float  v8f   __attribute__((ext_vector_type(8)));
typedef unsigned int v4u  __attribute__((ext_vector_type(4)));
typedef int          v8i_ __attribute__((ext_vector_type(8)));
typedef int          v4i_ __attribute__((ext_vector_type(4)));

// ---------------- problem constants ----------------
constexpr int kD   = 1024;    // model dim
constexpr int kH   = 4096;    // hidden dim
constexpr int kE   = 8;       // experts
constexpr int kN   = 8192;    // B*S tokens
constexpr int kNK  = kN * 2;  // top-2 assignments
constexpr int kCAP = 2151;    // ceil(N*K/E * 1.05)
constexpr int kMPAD = 2176;   // cap padded to 128 (17 M-tiles)

// ---------------- workspace layout (bytes) ----------------
constexpr size_t OFF_EIDX = 0;
constexpr size_t OFF_GATE = OFF_EIDX + (size_t)kNK * 4;
constexpr size_t OFF_DEST = OFF_GATE + (size_t)kNK * 4;
constexpr size_t OFF_WGT  = OFF_DEST + (size_t)kNK * 4;
constexpr size_t OFF_ABUF = OFF_WGT  + (size_t)kNK * 4;                 // bf16 [E*MPAD, D]
constexpr size_t OFF_W1T  = OFF_ABUF + (size_t)kE * kMPAD * kD * 2;     // bf16 [E, H, D]  (W1 transposed)
constexpr size_t OFF_W2T  = OFF_W1T  + (size_t)kE * kD * kH * 2;        // bf16 [E, D, H]  (W2 transposed)
constexpr size_t OFF_HBUF = OFF_W2T  + (size_t)kE * kH * kD * 2;        // bf16 [E*MPAD, H]
constexpr size_t OFF_YBUF = OFF_HBUF + (size_t)kE * kMPAD * kH * 2;     // f32  [E*MPAD+1, D]

// ---------------- small utility kernels ----------------
__global__ void zero_u4_kernel(uint4* __restrict__ p, long n4) {
    uint4 z; z.x = z.y = z.z = z.w = 0u;
    for (long i = (long)blockIdx.x * blockDim.x + threadIdx.x; i < n4;
         i += (long)gridDim.x * blockDim.x)
        p[i] = z;
}

// convert f32 -> bf16 WITH transpose: src [E][R][C] -> dst [E][C][R]
__global__ __launch_bounds__(256) void cvtT_kernel(const float* __restrict__ src,
                                                   __bf16* __restrict__ dst,
                                                   int R, int C) {
    __shared__ float tile[32][33];          // +1 pad to dodge bank conflicts
    const int e = blockIdx.z;
    src += (size_t)e * R * C;
    dst += (size_t)e * R * C;
    const int c0 = blockIdx.x * 32, r0 = blockIdx.y * 32;
    const int tx = threadIdx.x & 31, ty = threadIdx.x >> 5;   // ty: 0..7
#pragma unroll
    for (int i = 0; i < 4; ++i) {
        const int r = ty * 4 + i;
        tile[r][tx] = src[(size_t)(r0 + r) * C + c0 + tx];
    }
    __syncthreads();
#pragma unroll
    for (int i = 0; i < 4; ++i) {
        const int c = ty * 4 + i;
        dst[(size_t)(c0 + c) * R + r0 + tx] = (__bf16)tile[tx][c];
    }
}

// ---------------- gating: one wave (32 lanes) per token ----------------
__global__ __launch_bounds__(256) void gating_kernel(
    const float* __restrict__ x, const float* __restrict__ Wg,
    int* __restrict__ eidx, float* __restrict__ gate) {
    const int wave = threadIdx.x >> 5;
    const int lane = threadIdx.x & 31;
    const int t = blockIdx.x * 8 + wave;

    float acc[kE];
#pragma unroll
    for (int e = 0; e < kE; ++e) acc[e] = 0.f;

    const float* xr = x + (size_t)t * kD;
    for (int d = lane; d < kD; d += 32) {
        float xv = xr[d];
        const float4* w4 = (const float4*)(Wg + (size_t)d * kE);
        float4 wa = w4[0], wb = w4[1];
        acc[0] += xv * wa.x; acc[1] += xv * wa.y;
        acc[2] += xv * wa.z; acc[3] += xv * wa.w;
        acc[4] += xv * wb.x; acc[5] += xv * wb.y;
        acc[6] += xv * wb.z; acc[7] += xv * wb.w;
    }
#pragma unroll
    for (int e = 0; e < kE; ++e)
        for (int off = 16; off; off >>= 1)
            acc[e] += __shfl_xor(acc[e], off, 32);

    if (lane == 0) {
        int e0 = 0;
#pragma unroll
        for (int e = 1; e < kE; ++e) if (acc[e] > acc[e0]) e0 = e;
        int e1 = (e0 == 0) ? 1 : 0;
#pragma unroll
        for (int e = 0; e < kE; ++e)
            if (e != e0 && acc[e] > acc[e1]) e1 = e;
        float g1 = __expf(acc[e1] - acc[e0]);
        float inv = 1.f / (1.f + g1);
        eidx[2 * t]     = e0;  gate[2 * t]     = inv;
        eidx[2 * t + 1] = e1;  gate[2 * t + 1] = g1 * inv;
    }
}

// ---------------- routing: single wave, deterministic token-major order ----
__global__ void route_kernel(const int* __restrict__ eidx,
                             const float* __restrict__ gate,
                             int* __restrict__ dest, float* __restrict__ wgt) {
    const int lane = threadIdx.x;             // launched with exactly 32 threads
    const unsigned lmask = (1u << lane) - 1u;
    int cnt[kE];
#pragma unroll
    for (int e = 0; e < kE; ++e) cnt[e] = 0;

    for (int base = 0; base < kNK; base += 32) {
        const int i = base + lane;
        const int my_e = eidx[i];
        int pos = 0;
#pragma unroll
        for (int e = 0; e < kE; ++e) {
            unsigned m = (unsigned)__ballot(my_e == e);
            if (my_e == e) pos = cnt[e] + __popc(m & lmask);
            cnt[e] += __popc(m);
        }
        const bool keep = pos < kCAP;
        dest[i] = keep ? my_e * kMPAD + pos : kE * kMPAD;  // overflow -> safe row
        wgt[i]  = keep ? gate[i] : 0.f;
    }
}

// ---------------- scatter: token row (f32) -> expert buffer (bf16) --------
__global__ void scatter_kernel(const float* __restrict__ x,
                               const int* __restrict__ dest,
                               __bf16* __restrict__ Abuf) {
    const int i = blockIdx.x;
    const int slot = dest[i];
    if (slot >= kE * kMPAD) return;
    const float* src = x + (size_t)(i >> 1) * kD;
    __bf16* dst = Abuf + (size_t)slot * kD;
    for (int j = threadIdx.x; j < kD; j += blockDim.x)
        dst[j] = (__bf16)src[j];
}

// ---------------- TDM tile load: 128 rows x 64B, row stride Kd*2 bytes ----
// D# per CDNA5 ISA: count=1, type=2(image),
// data_size=3 (8B units), tile = 8 units x 128 rows, contiguous LDS dest.
__device__ __forceinline__ void tdm_load_tile(unsigned lds_addr,
                                              unsigned long long gaddr,
                                              int kd_units /* = Kd/4 */) {
    const unsigned t0 = (unsigned)kd_units;   // tensor_dim0 = full row in units
    const unsigned t1 = 1u << 20;             // tensor_dim1: plenty of rows
    v4u g0;
    g0[0] = 1u;                                                   // count=1, user mode
    g0[1] = lds_addr;                                             // LDS byte address
    g0[2] = (unsigned)gaddr;                                      // global_addr[31:0]
    g0[3] = (unsigned)((gaddr >> 32) & 0x01ffffffu) | (2u << 30); // addr[56:32] | type=2
    v8i_ g1;
    g1[0] = (int)(3u << 16);                                      // data_size=8B, mask=0
    g1[1] = (int)((t0 & 0xffffu) << 16);                          // tensor_dim0 lo16
    g1[2] = (int)(((t0 >> 16) & 0xffffu) | ((t1 & 0xffffu) << 16));
    g1[3] = (int)(((t1 >> 16) & 0xffffu) | (8u << 16));           // tile_dim0 = 8 units
    g1[4] = 128;                                                  // tile_dim1 = 128 rows
    g1[5] = (int)t0;                                              // tensor_dim0_stride lo32
    g1[6] = 0;
    g1[7] = 0;
    v4i_ z4 = {};
    v8i_ z8 = {};
    // 6-arg form (this toolchain): (g0, g1, g2, g3, g4, cpol)
    __builtin_amdgcn_tensor_load_to_lds(g0, g1, z4, z4, z8, 0);
}

// ---------------- WMMA bf16 GEMM: block 128x128, 8 waves x (64x32) --------
__device__ __forceinline__ float gelu_tanh(float v) {
    float c = 0.7978845608028654f * (v + 0.044715f * v * v * v);
    return 0.5f * v * (1.f + tanhf(c));
}

union FragBF { uint4 u[2]; v16bf v; };

template <bool GELU_BF16_OUT>
__global__ __launch_bounds__(256) void gemm_bf16_kernel(
    const __bf16* __restrict__ A,    // [E, MPAD, Kd] bf16 row-major
    const __bf16* __restrict__ Bt,   // [E, Nd, Kd]  bf16 row-major (pre-transposed weights)
    const float*  __restrict__ bias, // [E, Nd]
    void* __restrict__ Out,          // bf16 [E,MPAD,Nd] (GELU) or f32
    int Kd, int Nd) {
    __shared__ __bf16 As[2][128 * 32];
    __shared__ __bf16 Bs[2][128 * 32];

    const int tid  = threadIdx.x;
    const int wave = tid >> 5;
    const int lane = tid & 31;
    const int wm = wave >> 2;                 // 0..1 -> 64-row slab
    const int wn = wave & 3;                  // 0..3 -> 32-col slab
    const int laneLo = lane & 15;
    const int laneHi = lane >> 4;

    const int e  = blockIdx.z;
    const int m0 = blockIdx.y * 128;
    const int n0 = blockIdx.x * 128;

    const __bf16* Ag = A  + (size_t)e * kMPAD * Kd + (size_t)m0 * Kd;
    const __bf16* Bg = Bt + (size_t)e * Nd * Kd + (size_t)n0 * Kd;
    const int kdu = Kd >> 2;                  // row stride in 8-byte units

    const unsigned aLds[2] = { (unsigned)(size_t)&As[0][0], (unsigned)(size_t)&As[1][0] };
    const unsigned bLds[2] = { (unsigned)(size_t)&Bs[0][0], (unsigned)(size_t)&Bs[1][0] };

    v8f acc[4][2];
#pragma unroll
    for (int fa = 0; fa < 4; ++fa)
#pragma unroll
        for (int fb = 0; fb < 2; ++fb) acc[fa][fb] = (v8f){};

    const int numK = Kd / 32;

    // prologue: wave 0 kicks off the first pair of TDM tile loads
    if (wave == 0) {
        tdm_load_tile(aLds[0], (unsigned long long)(size_t)Ag, kdu);
        tdm_load_tile(bLds[0], (unsigned long long)(size_t)Bg, kdu);
    }

    for (int kt = 0; kt < numK; ++kt) {
        const int buf = kt & 1;
        // TENSORcnt is per-wave: wave 0 waits for its TDM pair; no-op for others.
        __builtin_amdgcn_s_wait_tensorcnt(0);
        asm volatile("" ::: "memory");
        // Single barrier per K-step: release compute on `buf` AND prove all
        // waves finished reading buf^1 (so TDM may overwrite it below).
        __syncthreads();
        if (wave == 0 && kt + 1 < numK) {
            const int k1 = (kt + 1) * 32;
            tdm_load_tile(aLds[buf ^ 1], (unsigned long long)(size_t)(Ag + k1), kdu);
            tdm_load_tile(bLds[buf ^ 1], (unsigned long long)(size_t)(Bg + k1), kdu);
        }

        // --- fragment loads (layouts per ISA 7.12.2) ---
        FragBF af[4];
#pragma unroll
        for (int fa = 0; fa < 4; ++fa) {
            const int row = wm * 64 + fa * 16 + laneLo;
            const int ho  = laneHi * 8;       // lanes>=16 hold K=8..15 / 24..31
            af[fa].u[0] = *(const uint4*)&As[buf][row * 32 + ho];
            af[fa].u[1] = *(const uint4*)&As[buf][row * 32 + ho + 16];
        }
        FragBF bfr[2];
#pragma unroll
        for (int fb = 0; fb < 2; ++fb) {
            const int col = wn * 32 + fb * 16 + laneLo;
            const int ko  = laneHi * 8;       // lanes>=16 hold K=8..15 / 24..31
            bfr[fb].u[0] = *(const uint4*)&Bs[buf][col * 32 + ko];
            bfr[fb].u[1] = *(const uint4*)&Bs[buf][col * 32 + ko + 16];
        }
#pragma unroll
        for (int fa = 0; fa < 4; ++fa)
#pragma unroll
            for (int fb = 0; fb < 2; ++fb)
                acc[fa][fb] = __builtin_amdgcn_wmma_f32_16x16x32_bf16(
                    false, af[fa].v, false, bfr[fb].v,
                    (short)0, acc[fa][fb], false, false);
    }

    // ---- epilogue ----
    const size_t outBase = (size_t)e * kMPAD * Nd;
#pragma unroll
    for (int fa = 0; fa < 4; ++fa) {
#pragma unroll
        for (int fb = 0; fb < 2; ++fb) {
            const int gn = n0 + wn * 32 + fb * 16 + laneLo;
            const float bv = bias[e * Nd + gn];
#pragma unroll
            for (int r = 0; r < 8; ++r) {
                const int gm = m0 + wm * 64 + fa * 16 + laneHi * 8 + r;
                float v = acc[fa][fb][r] + bv;
                if (GELU_BF16_OUT)
                    ((__bf16*)Out)[outBase + (size_t)gm * Nd + gn] = (__bf16)gelu_tanh(v);
                else
                    ((float*)Out)[outBase + (size_t)gm * Nd + gn] = v;
            }
        }
    }
}

// ---------------- gather-combine: out[t] = w0*y[d0] + w1*y[d1] -----------
__global__ void gather_kernel(const float* __restrict__ ybuf,
                              const int* __restrict__ dest,
                              const float* __restrict__ wgt,
                              float* __restrict__ out) {
    const int t = blockIdx.x;
    const int d0 = dest[2 * t], d1 = dest[2 * t + 1];
    const float w0 = wgt[2 * t], w1 = wgt[2 * t + 1];
    const float4* y0 = (const float4*)(ybuf + (size_t)d0 * kD);
    const float4* y1 = (const float4*)(ybuf + (size_t)d1 * kD);
    float4* o = (float4*)(out + (size_t)t * kD);
    for (int j = threadIdx.x; j < kD / 4; j += blockDim.x) {
        float4 a = y0[j], b = y1[j], r;
        r.x = w0 * a.x + w1 * b.x;
        r.y = w0 * a.y + w1 * b.y;
        r.z = w0 * a.z + w1 * b.z;
        r.w = w0 * a.w + w1 * b.w;
        o[j] = r;
    }
}

// ---------------- launch ----------------
extern "C" void kernel_launch(void* const* d_in, const int* in_sizes, int n_in,
                              void* d_out, int out_size, void* d_ws, size_t ws_size,
                              hipStream_t stream) {
    const float* x  = (const float*)d_in[0];
    const float* Wg = (const float*)d_in[1];
    const float* W1 = (const float*)d_in[2];
    const float* b1 = (const float*)d_in[3];
    const float* W2 = (const float*)d_in[4];
    const float* b2 = (const float*)d_in[5];
    float* out = (float*)d_out;

    char* ws = (char*)d_ws;
    int*    eidx = (int*)   (ws + OFF_EIDX);
    float*  gate = (float*) (ws + OFF_GATE);
    int*    dest = (int*)   (ws + OFF_DEST);
    float*  wgt  = (float*) (ws + OFF_WGT);
    __bf16* Abuf = (__bf16*)(ws + OFF_ABUF);
    __bf16* W1t  = (__bf16*)(ws + OFF_W1T);
    __bf16* W2t  = (__bf16*)(ws + OFF_W2T);
    __bf16* Hbuf = (__bf16*)(ws + OFF_HBUF);
    float*  ybuf = (float*) (ws + OFF_YBUF);

    // 1) zero padded expert-input buffer
    zero_u4_kernel<<<2048, 256, 0, stream>>>(
        (uint4*)Abuf, (long)kE * kMPAD * kD * 2 / 16);
    // 2) weights f32 -> bf16, transposed so GEMM B-tiles are contiguous rows
    cvtT_kernel<<<dim3(kH / 32, kD / 32, kE), 256, 0, stream>>>(W1, W1t, kD, kH);
    cvtT_kernel<<<dim3(kD / 32, kH / 32, kE), 256, 0, stream>>>(W2, W2t, kH, kD);
    // 3) gating
    gating_kernel<<<kN / 8, 256, 0, stream>>>(x, Wg, eidx, gate);
    // 4) deterministic capacity routing (single wave)
    route_kernel<<<1, 32, 0, stream>>>(eidx, gate, dest, wgt);
    // 5) scatter tokens into expert buffers (bf16)
    scatter_kernel<<<kNK, 256, 0, stream>>>(x, dest, Abuf);
    // 6) GEMM1 + fused GELU -> bf16 H
    gemm_bf16_kernel<true><<<dim3(kH / 128, kMPAD / 128, kE), 256, 0, stream>>>(
        Abuf, W1t, b1, (void*)Hbuf, kD, kH);
    // 7) GEMM2 + bias -> f32 y
    gemm_bf16_kernel<false><<<dim3(kD / 128, kMPAD / 128, kE), 256, 0, stream>>>(
        Hbuf, W2t, b2, (void*)ybuf, kH, kD);
    // 8) weighted gather-combine
    gather_kernel<<<kN, 256, 0, stream>>>(ybuf, dest, wgt, out);
}
